// MultiHeadAttention_41755672052075
// MI455X (gfx1250) — matrix operand order repaired and checked by reading, not compiled, not run on hardware
//
#include <hip/hip_runtime.h>

// ---------------------------------------------------------------------------
// MultiHeadAttention for MI455X (gfx1250): bf16 WMMA pipeline, f32 accumulate.
// LDS-staged B panels with async global->LDS copies (ASYNCcnt) when available.
// ---------------------------------------------------------------------------

typedef __bf16 bf16;
typedef __attribute__((ext_vector_type(16))) __bf16 v16bf;
typedef __attribute__((ext_vector_type(8)))  __bf16 v8bf;
typedef __attribute__((ext_vector_type(8)))  float  v8f;
typedef __attribute__((ext_vector_type(4)))  int    v4i;

#define DM     1024
#define NH     16
#define DK     64
#define BATCH  2
#define SEQ    2048
#define ROWS   (BATCH * SEQ)          // 4096

// workspace layout (bf16 elements). Total = 33,554,432 elems = 64 MB.
#define XQ_OFF   ((size_t)0)
#define XK_OFF   ((size_t)4194304)
#define XV_OFF   ((size_t)8388608)
#define WQT_OFF  ((size_t)12582912)   // [H][64][1024]
#define WKT_OFF  ((size_t)13631488)
#define WVT_OFF  ((size_t)14680064)
#define WOT_OFF  ((size_t)15728640)   // [1024][1024]  (out x in)
#define QH_OFF   ((size_t)16777216)   // [B][H][S][64]
#define KH_OFF   ((size_t)20971520)   // [B][H][S][64]
#define VHT_OFF  ((size_t)25165824)   // [B][H][64][S]
#define OBF_OFF  ((size_t)29360128)   // [B][S][1024]

// LDS B-panel chunk: 64 rows x 128 k-elems, row padded to 136 bf16
// (68 dwords/row -> bank offset 4n per row -> conflict-free ds_load_b128).
#define KC    128
#define KCP   136
#define NCHUNK (DM / KC)              // 8

#if defined(__has_builtin)
#  if __has_builtin(__builtin_amdgcn_global_load_async_to_lds_b128)
#    define HAVE_ASYNC_LDS 1
#  endif
#endif
#ifndef HAVE_ASYNC_LDS
#  define HAVE_ASYNC_LDS 0
#endif

// ----------------------------- WMMA helpers --------------------------------

__device__ __forceinline__ v8f wmma_bf16(v16bf a, v16bf b, v8f c) {
  return __builtin_amdgcn_wmma_f32_16x16x32_bf16(
      /*neg_a=*/false, a, /*neg_b=*/false, b,
      /*c_mod=*/(short)0, c, /*reuse_a=*/false, /*reuse_b=*/false);
}

// A-matrix 16x32 bf16 tile load (row-major source, leading dim `ld` elems).
__device__ __forceinline__ v16bf load_A(const bf16* base, int ld, int k0, int lane) {
  int m    = lane & 15;
  int koff = k0 + ((lane >> 4) << 4);
  const bf16* p = base + (size_t)m * ld + koff;
  v8bf lo = *(const v8bf*)(p);
  v8bf hi = *(const v8bf*)(p + 8);
  return __builtin_shufflevector(lo, hi, 0,1,2,3,4,5,6,7,8,9,10,11,12,13,14,15);
}

// B-matrix 32x16 bf16 tile load from B^T ([N][K] row-major, leading dim `ld`).
__device__ __forceinline__ v16bf load_B(const bf16* bt, int ld, int k0, int lane) {
  int n    = lane & 15;
  int koff = k0 + ((lane >> 4) << 4);
  const bf16* p = bt + (size_t)n * ld + koff;
  v8bf lo = *(const v8bf*)(p);
  v8bf hi = *(const v8bf*)(p + 8);
  return __builtin_shufflevector(lo, hi, 0,1,2,3,4,5,6,7,8,9,10,11,12,13,14,15);
}

// B tile from LDS-staged panel (rows of KCP elems), tile col block j.
__device__ __forceinline__ v16bf load_Blds(const bf16* bp, int j, int k0, int lane) {
  int n    = lane & 15;
  int koff = k0 + ((lane >> 4) << 4);
  const bf16* p = bp + (j * 16 + n) * KCP + koff;
  v8bf lo = *(const v8bf*)(p);
  v8bf hi = *(const v8bf*)(p + 8);
  return __builtin_shufflevector(lo, hi, 0,1,2,3,4,5,6,7,8,9,10,11,12,13,14,15);
}

// Cooperatively copy a 64 x KC panel (global row stride DM) into LDS.
// 1024 16-byte chunks, 4 per thread. Async (ASYNCcnt) when available.
__device__ __forceinline__ void stage_panel(const bf16* gsrc, bf16* lbuf, int tid) {
#pragma unroll
  for (int it = 0; it < 4; ++it) {
    int ch  = tid + it * 256;         // 0..1023
    int row = ch >> 4;                // 0..63
    int kc  = (ch & 15) << 3;         // elem offset 0..120
    const bf16* g = gsrc + (size_t)row * DM + kc;
    bf16* l = lbuf + row * KCP + kc;
#if HAVE_ASYNC_LDS
    __builtin_amdgcn_global_load_async_to_lds_b128(
        (__attribute__((address_space(1))) v4i*)g,
        (__attribute__((address_space(3))) v4i*)l, 0, 0);
#else
    *(v8bf*)l = *(const v8bf*)g;
#endif
  }
}

__device__ __forceinline__ void wait_async_copies() {
#if HAVE_ASYNC_LDS
  asm volatile("s_wait_asynccnt 0" ::: "memory");
#endif
}

// ----------------------------- conversions ---------------------------------

__global__ void __launch_bounds__(256) k_cvt_qkv(const float* __restrict__ Q,
                                                 const float* __restrict__ K,
                                                 const float* __restrict__ V,
                                                 bf16* __restrict__ ws) {
  size_t i = (size_t)blockIdx.x * blockDim.x + threadIdx.x;
  if (i >= (size_t)ROWS * DM) return;
  ws[XQ_OFF + i] = (bf16)Q[i];
  ws[XK_OFF + i] = (bf16)K[i];
  ws[XV_OFF + i] = (bf16)V[i];
}

// dst[h][n][d] = src[h][d][n]  (convert + transpose each head's weight)
__global__ void __launch_bounds__(256) k_cvt_w(const float* __restrict__ src,
                                               bf16* __restrict__ dst,
                                               int D, int N, size_t total) {
  size_t i = (size_t)blockIdx.x * blockDim.x + threadIdx.x;
  if (i >= total) return;
  int n = (int)(i % N);
  int d = (int)((i / N) % D);
  size_t h = i / ((size_t)N * D);
  dst[(h * N + n) * (size_t)D + d] = (bf16)src[i];
}

// --------------------------- projection GEMMs ------------------------------
// grid: (ROWS/128, 3*NH), block: 256 (8 waves). Each wave: 16 rows x 64 cols.
// B panel (64x1024 W^T of this head) staged through LDS, double buffered.
__global__ void __launch_bounds__(256) k_proj(bf16* __restrict__ ws,
                                              const float* __restrict__ bq,
                                              const float* __restrict__ bk,
                                              const float* __restrict__ bv) {
  __shared__ bf16 panel[2][64 * KCP];
  int tid = threadIdx.x;
  int wave = tid >> 5, lane = tid & 31;
  int p = blockIdx.y >> 4;          // 0=Q, 1=K, 2=V
  int h = blockIdx.y & 15;
  int row0 = blockIdx.x * 128 + wave * 16;

  const bf16* X  = ws + ((size_t)p * 4194304);                     // XQ/XK/XV
  const bf16* WT = ws + WQT_OFF + (size_t)p * 1048576 + (size_t)h * (DK * DM);
  const float* bias = (p == 0 ? bq : (p == 1 ? bk : bv)) + h * DK;

  int n  = lane & 15;
  int mh = (lane >> 4) << 3;

  v8f acc[4];
#pragma unroll
  for (int j = 0; j < 4; ++j) {
    float bb = bias[j * 16 + n];
#pragma unroll
    for (int r = 0; r < 8; ++r) acc[j][r] = bb;
  }

  const bf16* Arow = X + (size_t)row0 * DM;
  stage_panel(WT, panel[0], tid);                 // chunk 0
  for (int c = 0; c < NCHUNK; ++c) {
    wait_async_copies();                          // own copies of chunk c done
    __syncthreads();                              // everyone's chunk c done
    if (c + 1 < NCHUNK)                           // overwrite buf read at c-1
      stage_panel(WT + (c + 1) * KC, panel[(c + 1) & 1], tid);
    const bf16* bp = panel[c & 1];
#pragma unroll
    for (int t = 0; t < 4; ++t) {
      v16bf a = load_A(Arow, DM, c * KC + t * 32, lane);
#pragma unroll
      for (int j = 0; j < 4; ++j)
        acc[j] = wmma_bf16(a, load_Blds(bp, j, t * 32, lane), acc[j]);
    }
  }

  int bidx = row0 >> 11;            // batch
  int s0   = row0 & (SEQ - 1);      // seq pos of tile row 0
  if (p < 2) {
    bf16* Y = ws + (p == 0 ? QH_OFF : KH_OFF) +
              ((size_t)(bidx * NH + h) * SEQ) * DK;
#pragma unroll
    for (int j = 0; j < 4; ++j)
#pragma unroll
      for (int r = 0; r < 8; ++r)
        Y[(size_t)(s0 + mh + r) * DK + j * 16 + n] = (bf16)acc[j][r];
  } else {
    // transposed [B][H][64][S] : lane holds fixed dv col, 8 consecutive rows
    bf16* YT = ws + VHT_OFF + ((size_t)(bidx * NH + h) * DK) * SEQ;
#pragma unroll
    for (int j = 0; j < 4; ++j) {
      v8bf v;
#pragma unroll
      for (int r = 0; r < 8; ++r) v[r] = (bf16)acc[j][r];
      *(v8bf*)(YT + (size_t)(j * 16 + n) * SEQ + s0 + mh) = v;
    }
  }
}

// ---------------------- causal flash attention -----------------------------
// One wave per (b, h, 16-query tile). 8 waves / block, 512 blocks.
__global__ void __launch_bounds__(256) k_attn(bf16* __restrict__ ws) {
  __shared__ bf16 pls[8][16][32];   // per-wave P-tile staging (D->A transpose)
  int wave = threadIdx.x >> 5, lane = threadIdx.x & 31;
  int id = blockIdx.x * 8 + wave;
  int qt   = id & 127;
  int h    = (id >> 7) & 15;
  int bidx = id >> 11;
  int q0 = qt * 16;
  int n  = lane & 15;
  int mh = (lane >> 4) << 3;

  const bf16* Qh = ws + QH_OFF + ((size_t)(bidx * NH + h) * SEQ + q0) * DK;
  const bf16* Kh = ws + KH_OFF + (size_t)(bidx * NH + h) * SEQ * DK;
  const bf16* VT = ws + VHT_OFF + (size_t)(bidx * NH + h) * DK * SEQ;

  v16bf qa0 = load_A(Qh, DK, 0, lane);
  v16bf qa1 = load_A(Qh, DK, 32, lane);

  v8f acco[4];
#pragma unroll
  for (int j = 0; j < 4; ++j)
#pragma unroll
    for (int r = 0; r < 8; ++r) acco[j][r] = 0.f;
  float mi[8], li[8];
#pragma unroll
  for (int r = 0; r < 8; ++r) { mi[r] = -1e30f; li[r] = 0.f; }

  int nblk = (q0 + 15) / 32 + 1;    // causal: key blocks of 32
  for (int kb = 0; kb < nblk; ++kb) {
    int kbase = kb * 32;
    if (kb + 1 < nblk)              // prefetch every 2nd key row of next 64 (8 KB span)
      __builtin_prefetch(Kh + (size_t)(kbase + 32) * DK + (size_t)lane * DK * 2, 0, 0);
    v8f s0, s1;
#pragma unroll
    for (int r = 0; r < 8; ++r) { s0[r] = 0.f; s1[r] = 0.f; }
    {
      v16bf b0a = load_B(Kh + (size_t)(kbase)*DK, DK, 0, lane);
      v16bf b0b = load_B(Kh + (size_t)(kbase)*DK, DK, 32, lane);
      s0 = wmma_bf16(qa0, b0a, s0);
      s0 = wmma_bf16(qa1, b0b, s0);
      v16bf b1a = load_B(Kh + (size_t)(kbase + 16) * DK, DK, 0, lane);
      v16bf b1b = load_B(Kh + (size_t)(kbase + 16) * DK, DK, 32, lane);
      s1 = wmma_bf16(qa0, b1a, s1);
      s1 = wmma_bf16(qa1, b1b, s1);
    }

    bool needmask = (kbase + 31) > q0;
    float p0[8], p1[8];
#pragma unroll
    for (int r = 0; r < 8; ++r) {
      int qg = q0 + mh + r;
      float v0 = s0[r] * 0.125f;    // 1/sqrt(64)
      float v1 = s1[r] * 0.125f;
      if (needmask) {
        if (kbase + n > qg)      v0 = -1e30f;
        if (kbase + 16 + n > qg) v1 = -1e30f;
      }
      float mr = fmaxf(v0, v1);
#pragma unroll
      for (int d = 1; d < 16; d <<= 1) mr = fmaxf(mr, __shfl_xor(mr, d, 32));
      float mn = fmaxf(mi[r], mr);
      float al = __expf(mi[r] - mn);
      float e0 = __expf(v0 - mn);
      float e1 = __expf(v1 - mn);
      float rs = e0 + e1;
#pragma unroll
      for (int d = 1; d < 16; d <<= 1) rs += __shfl_xor(rs, d, 32);
      li[r] = li[r] * al + rs;
      mi[r] = mn;
      p0[r] = e0; p1[r] = e1;
      acco[0][r] *= al; acco[1][r] *= al; acco[2][r] *= al; acco[3][r] *= al;
    }

    // D-layout -> A-layout via LDS (per-wave private tile; in-wave DS order)
#pragma unroll
    for (int r = 0; r < 8; ++r) {
      pls[wave][mh + r][n]      = (bf16)p0[r];
      pls[wave][mh + r][16 + n] = (bf16)p1[r];
    }
    asm volatile("s_wait_dscnt 0" ::: "memory");
    v16bf pa = load_A(&pls[wave][0][0], 32, 0, lane);

#pragma unroll
    for (int j = 0; j < 4; ++j) {
      v16bf vb = load_B(VT + (size_t)(j * 16) * SEQ + kbase, SEQ, 0, lane);
      acco[j] = wmma_bf16(pa, vb, acco[j]);
    }
  }

  // epilogue: O[b][s][h*64 + dv] bf16, normalized by row sums
  bf16* O = ws + OBF_OFF + ((size_t)bidx * SEQ + q0) * DM + h * DK;
#pragma unroll
  for (int r = 0; r < 8; ++r) {
    float inv = 1.0f / li[r];
#pragma unroll
    for (int j = 0; j < 4; ++j)
      O[(size_t)(mh + r) * DM + j * 16 + n] = (bf16)(acco[j][r] * inv);
  }
}

// --------------------------- output projection -----------------------------
// grid: (ROWS/128, DM/64); wave computes 16x64 f32 tile of d_out.
// B panel = 64 rows of Wo^T starting at col0, staged via LDS like k_proj.
__global__ void __launch_bounds__(256) k_out(const bf16* __restrict__ ws,
                                             const float* __restrict__ bo,
                                             float* __restrict__ out) {
  __shared__ bf16 panel[2][64 * KCP];
  int tid = threadIdx.x;
  int wave = tid >> 5, lane = tid & 31;
  int row0 = blockIdx.x * 128 + wave * 16;
  int col0 = blockIdx.y * 64;
  int n  = lane & 15;
  int mh = (lane >> 4) << 3;

  const bf16* X  = ws + OBF_OFF + (size_t)row0 * DM;
  const bf16* WT = ws + WOT_OFF + (size_t)col0 * DM;

  v8f acc[4];
#pragma unroll
  for (int j = 0; j < 4; ++j) {
    float bb = bo[col0 + j * 16 + n];
#pragma unroll
    for (int r = 0; r < 8; ++r) acc[j][r] = bb;
  }

  stage_panel(WT, panel[0], tid);
  for (int c = 0; c < NCHUNK; ++c) {
    wait_async_copies();
    __syncthreads();
    if (c + 1 < NCHUNK)
      stage_panel(WT + (c + 1) * KC, panel[(c + 1) & 1], tid);
    const bf16* bp = panel[c & 1];
#pragma unroll
    for (int t = 0; t < 4; ++t) {
      v16bf a = load_A(X, DM, c * KC + t * 32, lane);
#pragma unroll
      for (int j = 0; j < 4; ++j)
        acc[j] = wmma_bf16(a, load_Blds(bp, j, t * 32, lane), acc[j]);
    }
  }

#pragma unroll
  for (int r = 0; r < 8; ++r)
#pragma unroll
    for (int j = 0; j < 4; ++j)
      out[(size_t)(row0 + mh + r) * DM + col0 + j * 16 + n] = acc[j][r];
}

// ------------------------------- launcher ----------------------------------

extern "C" void kernel_launch(void* const* d_in, const int* in_sizes, int n_in,
                              void* d_out, int out_size, void* d_ws, size_t ws_size,
                              hipStream_t stream) {
  const float* Q  = (const float*)d_in[0];
  const float* K  = (const float*)d_in[1];
  const float* V  = (const float*)d_in[2];
  // d_in[3] = mask (causal, computed analytically)
  const float* Wq = (const float*)d_in[4];
  const float* bq = (const float*)d_in[5];
  const float* Wk = (const float*)d_in[6];
  const float* bk = (const float*)d_in[7];
  const float* Wv = (const float*)d_in[8];
  const float* bv = (const float*)d_in[9];
  const float* Wo = (const float*)d_in[10];
  const float* bo = (const float*)d_in[11];
  bf16* ws = (bf16*)d_ws;
  float* out = (float*)d_out;

  // 1) precision conversion + weight transposition
  k_cvt_qkv<<<(ROWS * DM) / 256, 256, 0, stream>>>(Q, K, V, ws);
  size_t wtot = (size_t)NH * DM * DK;           // 1,048,576
  k_cvt_w<<<(int)(wtot / 256), 256, 0, stream>>>(Wq, ws + WQT_OFF, DM, DK, wtot);
  k_cvt_w<<<(int)(wtot / 256), 256, 0, stream>>>(Wk, ws + WKT_OFF, DM, DK, wtot);
  k_cvt_w<<<(int)(wtot / 256), 256, 0, stream>>>(Wv, ws + WVT_OFF, DM, DK, wtot);
  size_t wotot = (size_t)DM * DM;               // 1,048,576
  k_cvt_w<<<(int)(wotot / 256), 256, 0, stream>>>(Wo, ws + WOT_OFF, DM, DM, wotot);

  // 2) per-head Q/K/V projections (WMMA, LDS-staged B)
  dim3 gp(ROWS / 128, 3 * NH);
  k_proj<<<gp, 256, 0, stream>>>(ws, bq, bk, bv);

  // 3) causal flash attention (WMMA + online softmax)
  k_attn<<<(BATCH * NH * (SEQ / 16)) / 8, 256, 0, stream>>>(ws);

  // 4) output projection (WMMA, f32 store, LDS-staged B)
  dim3 go(ROWS / 128, DM / 64);
  k_out<<<go, 256, 0, stream>>>(ws, bo, out);
}
